// GIN_43997644981192
// MI455X (gfx1250) — matrix operand order repaired and checked by reading, not compiled, not run on hardware
//
#include <hip/hip_runtime.h>

typedef __attribute__((ext_vector_type(16))) __bf16 v16bf;
typedef __attribute__((ext_vector_type(8)))  __bf16 v8bf;
typedef __attribute__((ext_vector_type(8)))  float  v8f;

#define DIM     256
#define KTILES  8    // 256 / 32 (K per WMMA)
#define NTILES  16   // 256 / 16 (N tiles)

// ---------------------------------------------------------------------------
// Zero the aggregation workspace (grid-stride float4 stores).
// ---------------------------------------------------------------------------
__global__ void zero_f4(float4* __restrict__ p, long n4) {
    long i      = (long)blockIdx.x * blockDim.x + threadIdx.x;
    long stride = (long)gridDim.x * blockDim.x;
    float4 z = {0.f, 0.f, 0.f, 0.f};
    for (; i < n4; i += stride) p[i] = z;
}

// ---------------------------------------------------------------------------
// Pack a 256x256 f32 row-major weight matrix into bf16 WMMA B-fragments.
// B is 32x16 per tile; lane L (<16): col n = L, elems 0..15 -> K = kbase+0..15
// with kbase = kt*32; lanes 16..31: same cols, K = kt*32+16..31.
// Packed element (kt, nt, lane, e) stored contiguously per lane so the GEMM
// loads one aligned 32-byte v16bf per lane.
// ---------------------------------------------------------------------------
__global__ void pack_w_bf16(const float* __restrict__ W, __bf16* __restrict__ Wp) {
    int id = blockIdx.x * blockDim.x + threadIdx.x;   // 0 .. 4095
    if (id >= KTILES * NTILES * 32) return;
    int lane  = id & 31;
    int t     = id >> 5;
    int nt    = t & (NTILES - 1);
    int kt    = t >> 4;
    int n     = nt * 16 + (lane & 15);
    int kbase = kt * 32 + ((lane < 16) ? 0 : 16);
    __bf16* dst = Wp + (size_t)id * 16;
#pragma unroll
    for (int e = 0; e < 16; ++e)
        dst[e] = (__bf16)W[(size_t)(kbase + e) * DIM + n];
}

// ---------------------------------------------------------------------------
// Edge aggregation: aggr[row[e]] += val[e] * feat[col[e]].
// One wave32 per edge; lane owns 8 contiguous floats (2 x float4 gather,
// 8 x global_atomic_add_f32 scatter). feat+aggr (200MB) fit in the 192MB L2
// almost entirely, so this runs near L2 bandwidth. Prefetch next gather row.
// ---------------------------------------------------------------------------
__launch_bounds__(256)
__global__ void edge_aggregate(const float* __restrict__ feat,
                               const int*   __restrict__ erow,
                               const int*   __restrict__ ecol,
                               const float* __restrict__ eval_,
                               float*       __restrict__ aggr,
                               int nE) {
    const int  lane = threadIdx.x & 31;
    const long wid  = ((long)blockIdx.x * blockDim.x + threadIdx.x) >> 5;
    const long nw   = ((long)gridDim.x * blockDim.x) >> 5;
    for (long e = wid; e < nE; e += nw) {
        int   r = erow[e];
        int   c = ecol[e];
        float v = eval_[e];
        long en = e + nw;
        if (en < nE) {  // prefetch next edge's feature row (global_prefetch_b8)
            const char* pf = (const char*)(feat + (size_t)ecol[en] * DIM) + lane * 32;
            __builtin_prefetch(pf, 0, 1);
        }
        const float4* src = (const float4*)(feat + (size_t)c * DIM) + lane * 2;
        float4 f0 = src[0];
        float4 f1 = src[1];
        float* dst = aggr + (size_t)r * DIM + lane * 8;
        atomicAdd(dst + 0, v * f0.x);
        atomicAdd(dst + 1, v * f0.y);
        atomicAdd(dst + 2, v * f0.z);
        atomicAdd(dst + 3, v * f0.w);
        atomicAdd(dst + 4, v * f1.x);
        atomicAdd(dst + 5, v * f1.y);
        atomicAdd(dst + 6, v * f1.z);
        atomicAdd(dst + 7, v * f1.w);
    }
}

// ---------------------------------------------------------------------------
// Fused GIN MLP + norm: per block of 16 nodes,
//   A = aggr + (1+eps)*feat           (bf16, LDS)
//   h1 = relu(A @ W1 + b1)            (WMMA bf16 -> f32, bf16 back to LDS)
//   h2 = relu(h1 @ W2 + b2)           (WMMA bf16 -> f32, f32 to LDS)
//   out = (h2 - mean)*scale*rsqrt(var+1e-10) + offset   (wave reductions)
// 256 threads = 8 waves; wave w owns N-tiles {2w, 2w+1} (full N = 256).
// ---------------------------------------------------------------------------
__launch_bounds__(256)
__global__ void gin_mlp_fused(const float*  __restrict__ aggr,
                              const float*  __restrict__ feat,
                              const __bf16* __restrict__ W1p,
                              const float*  __restrict__ b1,
                              const __bf16* __restrict__ W2p,
                              const float*  __restrict__ b2,
                              const float*  __restrict__ epsp,
                              const float*  __restrict__ scale,
                              const float*  __restrict__ offset,
                              float*        __restrict__ out,
                              int nNodes) {
    __shared__ __align__(16) __bf16 Ash[16 * DIM];   // A (bf16)
    __shared__ __align__(16) __bf16 Hsh[16 * DIM];   // h1 (bf16)
    __shared__ __align__(16) float  H2 [16 * DIM];   // h2 (f32)

    const int tid  = threadIdx.x;
    const int lane = tid & 31;
    const int wave = tid >> 5;
    const int m0   = blockIdx.x * 16;
    const float se = 1.0f + epsp[0];

    // ---- stage A = aggr + (1+eps)*feat into LDS (row-major bf16) ----
#pragma unroll
    for (int i = 0; i < 16; ++i) {
        int node = m0 + i;
        float x = 0.f;
        if (node < nNodes) {
            size_t idx = (size_t)node * DIM + tid;
            x = aggr[idx] + se * feat[idx];
        }
        Ash[i * DIM + tid] = (__bf16)x;
    }
    __syncthreads();

    const int rowA = lane & 15;
    const int ksel = (lane < 16) ? 0 : 16;  // A-frag K-half select (contiguous: K = 16h + s)
    const int nt0  = wave * 2;
    const int nt1  = wave * 2 + 1;
    const int col0 = nt0 * 16 + (lane & 15);
    const int col1 = nt1 * 16 + (lane & 15);
    const int rb   = (lane < 16) ? 0 : 8;   // C/D row base per lane half

    v8f zero = {};
    v8f c0 = zero, c1 = zero;

    // ---- GEMM1: A @ W1 ----
    {
        const v16bf* Bp = (const v16bf*)W1p;
#pragma unroll
        for (int kt = 0; kt < KTILES; ++kt) {
            union { v16bf v; v8bf h[2]; } ua;
            const int abase = rowA * DIM + kt * 32 + ksel;
            ua.h[0] = *(const v8bf*)&Ash[abase];        // K = base+0..7
            ua.h[1] = *(const v8bf*)&Ash[abase + 8];    // K = base+8..15
            v16bf bb0 = Bp[(kt * NTILES + nt0) * 32 + lane];
            v16bf bb1 = Bp[(kt * NTILES + nt1) * 32 + lane];
            c0 = __builtin_amdgcn_wmma_f32_16x16x32_bf16(false, ua.v, false, bb0,
                                                         (short)0, c0, false, false);
            c1 = __builtin_amdgcn_wmma_f32_16x16x32_bf16(false, ua.v, false, bb1,
                                                         (short)0, c1, false, false);
        }
    }
    // ---- h1 = relu(c + b1) -> LDS bf16 (C/D layout scatter) ----
    {
        float bia0 = b1[col0], bia1 = b1[col1];
#pragma unroll
        for (int v = 0; v < 8; ++v) {
            float h0 = c0[v] + bia0; h0 = h0 > 0.f ? h0 : 0.f;
            float h1 = c1[v] + bia1; h1 = h1 > 0.f ? h1 : 0.f;
            Hsh[(rb + v) * DIM + col0] = (__bf16)h0;
            Hsh[(rb + v) * DIM + col1] = (__bf16)h1;
        }
    }
    __syncthreads();

    // ---- GEMM2: h1 @ W2 ----
    c0 = zero; c1 = zero;
    {
        const v16bf* Bp = (const v16bf*)W2p;
#pragma unroll
        for (int kt = 0; kt < KTILES; ++kt) {
            union { v16bf v; v8bf h[2]; } ua;
            const int abase = rowA * DIM + kt * 32 + ksel;
            ua.h[0] = *(const v8bf*)&Hsh[abase];
            ua.h[1] = *(const v8bf*)&Hsh[abase + 8];
            v16bf bb0 = Bp[(kt * NTILES + nt0) * 32 + lane];
            v16bf bb1 = Bp[(kt * NTILES + nt1) * 32 + lane];
            c0 = __builtin_amdgcn_wmma_f32_16x16x32_bf16(false, ua.v, false, bb0,
                                                         (short)0, c0, false, false);
            c1 = __builtin_amdgcn_wmma_f32_16x16x32_bf16(false, ua.v, false, bb1,
                                                         (short)0, c1, false, false);
        }
    }
    // ---- h2 = relu(c + b2) -> LDS f32 ----
    {
        float bia0 = b2[col0], bia1 = b2[col1];
#pragma unroll
        for (int v = 0; v < 8; ++v) {
            float h0 = c0[v] + bia0; h0 = h0 > 0.f ? h0 : 0.f;
            float h1 = c1[v] + bia1; h1 = h1 > 0.f ? h1 : 0.f;
            H2[(rb + v) * DIM + col0] = h0;
            H2[(rb + v) * DIM + col1] = h1;
        }
    }
    __syncthreads();

    // ---- per-node norm: wave w handles rows 2w and 2w+1 ----
#pragma unroll
    for (int rr = 0; rr < 2; ++rr) {
        int row  = wave * 2 + rr;
        int node = m0 + row;
        float s = 0.f, s2 = 0.f;
#pragma unroll
        for (int it = 0; it < 8; ++it) {
            float x = H2[row * DIM + lane + it * 32];
            s  += x;
            s2 += x * x;
        }
#pragma unroll
        for (int off = 16; off > 0; off >>= 1) {
            s  += __shfl_xor(s,  off, 32);
            s2 += __shfl_xor(s2, off, 32);
        }
        float mean = s * (1.0f / DIM);
        float var  = s2 * (1.0f / DIM) - mean * mean + 1e-10f;
        float rinv = rsqrtf(var);
        if (node < nNodes) {
#pragma unroll
            for (int it = 0; it < 8; ++it) {
                int j   = lane + it * 32;
                float x = H2[row * DIM + j];
                out[(size_t)node * DIM + j] = (x - mean) * scale[j] * rinv + offset[j];
            }
        }
    }
}

// ---------------------------------------------------------------------------
extern "C" void kernel_launch(void* const* d_in, const int* in_sizes, int n_in,
                              void* d_out, int out_size, void* d_ws, size_t ws_size,
                              hipStream_t stream) {
    const float* feat   = (const float*)d_in[0];
    const int*   erow   = (const int*)  d_in[1];
    const int*   ecol   = (const int*)  d_in[2];
    const float* eval_  = (const float*)d_in[3];
    const float* W1     = (const float*)d_in[4];
    const float* b1     = (const float*)d_in[5];
    const float* W2     = (const float*)d_in[6];
    const float* b2     = (const float*)d_in[7];
    const float* eps    = (const float*)d_in[8];
    const float* offset = (const float*)d_in[9];
    const float* scale  = (const float*)d_in[10];

    const int nNodes = in_sizes[0] / DIM;
    const int nE     = in_sizes[1];

    char*  ws        = (char*)d_ws;
    float* aggr      = (float*)ws;                              // nNodes*DIM f32
    size_t aggrBytes = (size_t)nNodes * DIM * sizeof(float);
    size_t wOff      = (aggrBytes + 255) & ~(size_t)255;
    __bf16* W1p = (__bf16*)(ws + wOff);                         // 128 KB
    __bf16* W2p = W1p + (size_t)DIM * DIM;                      // 128 KB
    (void)ws_size; (void)n_in; (void)out_size;

    long n4 = (long)nNodes * DIM / 4;
    zero_f4<<<2048, 256, 0, stream>>>((float4*)aggr, n4);
    pack_w_bf16<<<16, 256, 0, stream>>>(W1, W1p);
    pack_w_bf16<<<16, 256, 0, stream>>>(W2, W2p);
    edge_aggregate<<<4096, 256, 0, stream>>>(feat, erow, ecol, eval_, aggr, nE);

    int grid = (nNodes + 15) / 16;
    gin_mlp_fused<<<grid, 256, 0, stream>>>(aggr, feat, W1p, b1, W2p, b2,
                                            eps, scale, offset,
                                            (float*)d_out, nNodes);
}
